// HyperCHARM_10677288698624
// MI455X (gfx1250) — matrix operand order repaired and checked by reading, not compiled, not run on hardware
//
#include <hip/hip_runtime.h>

// ---------------------------------------------------------------------------
// HyperCHARM forward for gfx1250 (MI455X, wave32, WMMA).
// All large GEMMs use v_wmma_f32_16x16x32_bf16 (bf16 inputs, f32 accum).
// Fragments are loaded as 16-byte vectors; all A operands come from
// pre-converted bf16 buffers so the GEMM inner loops carry no cvt chains.
// ---------------------------------------------------------------------------

#define N_NODES   100000
#define N_EDGES   50000
#define DEG       8
#define INC_N     (N_NODES * DEG)     // 800000
#define NODE_DIM  64
#define HEDGE_DIM 32
#define HID       128
#define PRED_H    64
#define EPS_LN    1e-5f
#define EPS_DIV   1e-6f

typedef __bf16 v16bf __attribute__((ext_vector_type(16)));
typedef __bf16 v8bf  __attribute__((ext_vector_type(8)));
typedef float  v8f   __attribute__((ext_vector_type(8)));

__device__ __forceinline__ __bf16 f2bf(float f) {
  unsigned u = __builtin_bit_cast(unsigned, f);
  unsigned r = u + 0x7FFFu + ((u >> 16) & 1u);   // round-to-nearest-even
  unsigned short h = (unsigned short)(r >> 16);
  return __builtin_bit_cast(__bf16, h);
}

__device__ __forceinline__ v8f zero8() {
  v8f z = {0.f, 0.f, 0.f, 0.f, 0.f, 0.f, 0.f, 0.f};
  return z;
}

__device__ __forceinline__ v8f wmma_bf(v16bf a, v16bf b, v8f c) {
  return __builtin_amdgcn_wmma_f32_16x16x32_bf16(false, a, false, b, (short)0, c,
                                                 false, false);
}

// A fragment (16x32 bf16): lane holds row (lane&15); K runs
// [koff + (lane&16?8:0) .. +7] and the same +16 (per ISA 16-bit A layout).
// Two 16-byte vector loads, no per-element packing.
__device__ __forceinline__ v16bf frag_a_bf(const __bf16* rowptr, int lane, int koff) {
  int kb0 = koff + ((lane & 16) ? 8 : 0);
  v8bf lo = *(const v8bf*)(rowptr + kb0);
  v8bf hi = *(const v8bf*)(rowptr + kb0 + 16);
  return __builtin_shufflevector(lo, hi, 0, 1, 2, 3, 4, 5, 6, 7,
                                 8, 9, 10, 11, 12, 13, 14, 15);
}

// B fragment (32x16 bf16): weights pre-transposed so column n is K-contiguous.
// colptr = Wt + n*ldk; lane holds column (lane&15), K = koff + (lane&16?8:0)..+7 and the same +16.
__device__ __forceinline__ v16bf frag_b_bf(const __bf16* colptr, int lane, int koff) {
  int kb = koff + ((lane & 16) ? 8 : 0);
  v8bf lo = *(const v8bf*)(colptr + kb);
  v8bf hi = *(const v8bf*)(colptr + kb + 16);
  return __builtin_shufflevector(lo, hi, 0, 1, 2, 3, 4, 5, 6, 7,
                                 8, 9, 10, 11, 12, 13, 14, 15);
}

// Sum across a 16-lane half of the wave (xor masks stay inside the half).
__device__ __forceinline__ float half16_sum(float v) {
#pragma unroll
  for (int m = 1; m < 16; m <<= 1) v += __shfl_xor(v, m, 32);
  return v;
}

__device__ __forceinline__ float wave32_sum(float v) {
#pragma unroll
  for (int m = 1; m < 32; m <<= 1) v += __shfl_xor(v, m, 32);
  return v;
}

// ---------------------------------------------------------------------------
// Elementwise f32 -> bf16 conversion (x, he_attr prepass).
// ---------------------------------------------------------------------------
__global__ void k_cvt_bf16(const float* __restrict__ src, __bf16* __restrict__ dst,
                           int n) {
  int idx = blockIdx.x * blockDim.x + threadIdx.x;
  if (idx < n) dst[idx] = f2bf(src[idx]);
}

// ---------------------------------------------------------------------------
// Weight transpose + f32 -> bf16 : Wt[c*din + r] = W[r*dout + c]
// ---------------------------------------------------------------------------
__global__ void k_transpose_bf16(const float* __restrict__ W, __bf16* __restrict__ Wt,
                                 int din, int dout) {
  int idx = blockIdx.x * blockDim.x + threadIdx.x;
  if (idx >= din * dout) return;
  int c = idx % dout, r = idx / dout;
  Wt[(size_t)c * din + r] = f2bf(W[idx]);
}

// ---------------------------------------------------------------------------
// h = relu(x @ inW + b); also emit bf16 copy of h for later WMMA A-operands.
// Block: 256 thr = 8 waves, 128 rows; K = 64 (2 k-steps).
// ---------------------------------------------------------------------------
__global__ void k_in_proj(const __bf16* __restrict__ xbf, const __bf16* __restrict__ inWt,
                          const float* __restrict__ in_b,
                          float* __restrict__ h, __bf16* __restrict__ hbf) {
  int lane = threadIdx.x & 31, w = threadIdx.x >> 5;
  int row0 = blockIdx.x * 128 + w * 16;
  int arow = row0 + (lane & 15);
  if (arow >= N_NODES) arow = N_NODES - 1;
  const __bf16* xp = xbf + (size_t)arow * NODE_DIM;
  v16bf a0 = frag_a_bf(xp, lane, 0);
  v16bf a1 = frag_a_bf(xp, lane, 32);
  int colb = lane & 15, half = (lane & 16) ? 8 : 0;
  bool full = (row0 + 16 <= N_NODES);
#pragma unroll
  for (int ct = 0; ct < 8; ++ct) {
    int col = 16 * ct + colb;
    const __bf16* bp = inWt + (size_t)col * NODE_DIM;
    v8f acc = zero8();
    acc = wmma_bf(a0, frag_b_bf(bp, lane, 0), acc);
    acc = wmma_bf(a1, frag_b_bf(bp, lane, 32), acc);
    float bias = in_b[col];
    if (full) {
#pragma unroll
      for (int r = 0; r < 8; ++r) {
        int row = row0 + r + half;
        float v = fmaxf(acc[r] + bias, 0.f);
        h[(size_t)row * HID + col]   = v;
        hbf[(size_t)row * HID + col] = f2bf(v);
      }
    } else {
#pragma unroll
      for (int r = 0; r < 8; ++r) {
        int row = row0 + r + half;
        if (row < N_NODES) {
          float v = fmaxf(acc[r] + bias, 0.f);
          h[(size_t)row * HID + col]   = v;
          hbf[(size_t)row * HID + col] = f2bf(v);
        }
      }
    }
  }
}

// ---------------------------------------------------------------------------
// node2edge: msg = MLP1([h[node], mark[he]]); atomicAdd into aggf[he].
// Block handles 128 incidences (node = inc >> 3). Two WMMA GEMMs, K=128 each;
// the 2 mark columns are a rank-2 VALU correction on the accumulators.
// ---------------------------------------------------------------------------
__global__ void k_node2edge(const __bf16* __restrict__ hbf, const int* __restrict__ he_ids,
                            const float* __restrict__ he_mark,
                            const __bf16* __restrict__ W1t, const float* __restrict__ W1full,
                            const float* __restrict__ b1, const float* __restrict__ g1,
                            const float* __restrict__ bt1,
                            const __bf16* __restrict__ W2t, const float* __restrict__ b2,
                            float* __restrict__ aggf) {
  __shared__ __bf16 hidB[128 * 136];
  __shared__ int   sHe[128];
  __shared__ float sM0[128], sM1[128];
  int t = threadIdx.x, lane = t & 31, w = t >> 5;
  int i0 = blockIdx.x * 128;
  if (t < 128) {
    int he = he_ids[i0 + t];
    sHe[t] = he;
    sM0[t] = he_mark[2 * he];
    sM1[t] = he_mark[2 * he + 1];
  }
  __syncthreads();

  int arow_l = w * 16 + (lane & 15);
  int node = (i0 + arow_l) >> 3;
  const __bf16* ap = hbf + (size_t)node * HID;
  int colb = lane & 15, half = (lane & 16) ? 8 : 0;

  v8f acc[8];
#pragma unroll
  for (int ct = 0; ct < 8; ++ct) acc[ct] = zero8();
#pragma unroll
  for (int ks = 0; ks < 4; ++ks) {
    v16bf a = frag_a_bf(ap, lane, ks * 32);
#pragma unroll
    for (int ct = 0; ct < 8; ++ct) {
      int col = 16 * ct + colb;
      acc[ct] = wmma_bf(a, frag_b_bf(W1t + (size_t)col * 128, lane, ks * 32), acc[ct]);
    }
  }

  // bias + mark-columns correction, cache LN affine params
  float g1v[8], btv[8];
#pragma unroll
  for (int ct = 0; ct < 8; ++ct) {
    int col = 16 * ct + colb;
    float bb  = b1[col];
    float m0w = W1full[128 * 128 + col];
    float m1w = W1full[129 * 128 + col];
    g1v[ct] = g1[col];
    btv[ct] = bt1[col];
#pragma unroll
    for (int r = 0; r < 8; ++r) {
      int il = w * 16 + r + half;
      acc[ct][r] += bb + sM0[il] * m0w + sM1[il] * m1w;
    }
  }

  // LayerNorm over 128 cols + ReLU, write bf16 tile for GEMM2
#pragma unroll
  for (int r = 0; r < 8; ++r) {
    float s = 0.f, q = 0.f;
#pragma unroll
    for (int ct = 0; ct < 8; ++ct) { float v = acc[ct][r]; s += v; q += v * v; }
    s = half16_sum(s); q = half16_sum(q);
    float mean = s * (1.f / 128.f);
    float var  = q * (1.f / 128.f) - mean * mean;
    float rstd = rsqrtf(var + EPS_LN);
    int rowl = w * 16 + r + half;
#pragma unroll
    for (int ct = 0; ct < 8; ++ct) {
      int col = 16 * ct + colb;
      float v = fmaxf((acc[ct][r] - mean) * rstd * g1v[ct] + btv[ct], 0.f);
      hidB[rowl * 136 + col] = f2bf(v);
    }
  }
  __syncthreads();

  // GEMM2: hid @ W2 + b2, then scatter-add into aggf[he]
  const __bf16* ap2 = hidB + (size_t)(w * 16 + (lane & 15)) * 136;
  v8f acc2[8];
#pragma unroll
  for (int ct = 0; ct < 8; ++ct) acc2[ct] = zero8();
#pragma unroll
  for (int ks = 0; ks < 4; ++ks) {
    v16bf a = frag_a_bf(ap2, lane, ks * 32);
#pragma unroll
    for (int ct = 0; ct < 8; ++ct) {
      int col = 16 * ct + colb;
      acc2[ct] = wmma_bf(a, frag_b_bf(W2t + (size_t)col * 128, lane, ks * 32), acc2[ct]);
    }
  }
#pragma unroll
  for (int ct = 0; ct < 8; ++ct) {
    int col = 16 * ct + colb;
    float bb = b2[col];
#pragma unroll
    for (int r = 0; r < 8; ++r) {
      int il = w * 16 + r + half;
      atomicAdd(&aggf[(size_t)sHe[il] * HID + col], acc2[ct][r] + bb);
    }
  }
}

// ---------------------------------------------------------------------------
// aggbf = (aggf / (count + eps)) as bf16
// ---------------------------------------------------------------------------
__global__ void k_agg_div(const float* __restrict__ aggf, const float* __restrict__ he_count,
                          __bf16* __restrict__ aggbf) {
  size_t idx = (size_t)blockIdx.x * 256 + threadIdx.x;
  float c = he_count[idx >> 7];
  aggbf[idx] = f2bf(aggf[idx] / (c + EPS_DIV));
}

// ---------------------------------------------------------------------------
// edge2node: inc = relu(MLP2([attr[he], agg[he]])); per-node mean over the 8
// consecutive incidences; h += LN(mean). Block = 128 incidences = 16 nodes.
// K = 160 (1 attr k-step + 4 agg k-steps).
// ---------------------------------------------------------------------------
__global__ void k_edge2node(const __bf16* __restrict__ attrbf, const __bf16* __restrict__ aggbf,
                            const int* __restrict__ he_ids,
                            const __bf16* __restrict__ W1t, const float* __restrict__ b1,
                            const float* __restrict__ g1, const float* __restrict__ bt1,
                            const __bf16* __restrict__ W2t, const float* __restrict__ b2,
                            const float* __restrict__ lng, const float* __restrict__ lnb,
                            float* __restrict__ h, __bf16* __restrict__ hbf) {
  __shared__ __bf16 hidB[128 * 136];
  __shared__ int   sHe[128];
  __shared__ float nodeAcc[16 * 132];
  int t = threadIdx.x, lane = t & 31, w = t >> 5;
  int i0 = blockIdx.x * 128;
  if (t < 128) sHe[t] = he_ids[i0 + t];
  for (int k = t; k < 16 * 132; k += 256) nodeAcc[k] = 0.f;
  __syncthreads();

  int colb = lane & 15, half = (lane & 16) ? 8 : 0;
  int he_a = sHe[w * 16 + (lane & 15)];
  const __bf16* attr_p = attrbf + (size_t)he_a * HEDGE_DIM;
  const __bf16* agg_p  = aggbf + (size_t)he_a * HID;

  v8f acc[8];
#pragma unroll
  for (int ct = 0; ct < 8; ++ct) acc[ct] = zero8();
  {
    v16bf a = frag_a_bf(attr_p, lane, 0);   // K 0..31 = he_attr
#pragma unroll
    for (int ct = 0; ct < 8; ++ct) {
      int col = 16 * ct + colb;
      acc[ct] = wmma_bf(a, frag_b_bf(W1t + (size_t)col * 160, lane, 0), acc[ct]);
    }
  }
#pragma unroll
  for (int ks = 1; ks < 5; ++ks) {          // K 32..159 = agg_e
    v16bf a = frag_a_bf(agg_p, lane, ks * 32 - 32);
#pragma unroll
    for (int ct = 0; ct < 8; ++ct) {
      int col = 16 * ct + colb;
      acc[ct] = wmma_bf(a, frag_b_bf(W1t + (size_t)col * 160, lane, ks * 32), acc[ct]);
    }
  }

  float g1v[8], btv[8];
#pragma unroll
  for (int ct = 0; ct < 8; ++ct) {
    int col = 16 * ct + colb;
    float bb = b1[col];
    g1v[ct] = g1[col];
    btv[ct] = bt1[col];
#pragma unroll
    for (int r = 0; r < 8; ++r) acc[ct][r] += bb;
  }
#pragma unroll
  for (int r = 0; r < 8; ++r) {
    float s = 0.f, q = 0.f;
#pragma unroll
    for (int ct = 0; ct < 8; ++ct) { float v = acc[ct][r]; s += v; q += v * v; }
    s = half16_sum(s); q = half16_sum(q);
    float mean = s * (1.f / 128.f);
    float var  = q * (1.f / 128.f) - mean * mean;
    float rstd = rsqrtf(var + EPS_LN);
    int rowl = w * 16 + r + half;
#pragma unroll
    for (int ct = 0; ct < 8; ++ct) {
      int col = 16 * ct + colb;
      float v = fmaxf((acc[ct][r] - mean) * rstd * g1v[ct] + btv[ct], 0.f);
      hidB[rowl * 136 + col] = f2bf(v);
    }
  }
  __syncthreads();

  const __bf16* ap2 = hidB + (size_t)(w * 16 + (lane & 15)) * 136;
  v8f acc2[8];
#pragma unroll
  for (int ct = 0; ct < 8; ++ct) acc2[ct] = zero8();
#pragma unroll
  for (int ks = 0; ks < 4; ++ks) {
    v16bf a = frag_a_bf(ap2, lane, ks * 32);
#pragma unroll
    for (int ct = 0; ct < 8; ++ct) {
      int col = 16 * ct + colb;
      acc2[ct] = wmma_bf(a, frag_b_bf(W2t + (size_t)col * 128, lane, ks * 32), acc2[ct]);
    }
  }
#pragma unroll
  for (int ct = 0; ct < 8; ++ct) {
    int col = 16 * ct + colb;
    float bb = b2[col];
#pragma unroll
    for (int r = 0; r < 8; ++r) {
      float v = fmaxf(acc2[ct][r] + bb, 0.f);           // outer ReLU
      int nl = (w * 16 + r + half) >> 3;                // node within block
      atomicAdd(&nodeAcc[nl * 132 + col], v);           // LDS ds_add_f32
    }
  }
  __syncthreads();

  // per-node: out = sum/(deg+eps); h += LN(out). Wave w handles nodes 2w, 2w+1.
#pragma unroll
  for (int k = 0; k < 2; ++k) {
    int nr = 2 * w + k;
    int ng = (i0 >> 3) + nr;
    float vals[4], s = 0.f, q = 0.f;
#pragma unroll
    for (int j = 0; j < 4; ++j) {
      int c = lane + 32 * j;
      float v = nodeAcc[nr * 132 + c] * (1.f / ((float)DEG + EPS_DIV));
      vals[j] = v; s += v; q += v * v;
    }
    s = wave32_sum(s); q = wave32_sum(q);
    float mean = s * (1.f / 128.f);
    float var  = q * (1.f / 128.f) - mean * mean;
    float rstd = rsqrtf(var + EPS_LN);
    if (ng < N_NODES) {
#pragma unroll
      for (int j = 0; j < 4; ++j) {
        int c = lane + 32 * j;
        float hv = h[(size_t)ng * HID + c] + (vals[j] - mean) * rstd * lng[c] + lnb[c];
        h[(size_t)ng * HID + c]   = hv;
        hbf[(size_t)ng * HID + c] = f2bf(hv);
      }
    }
  }
}

// ---------------------------------------------------------------------------
// head: out[n] = relu(LN(h @ p1W + p1b)) . p2W + p2b   (PRED_H = 64)
// ---------------------------------------------------------------------------
__global__ void k_head(const __bf16* __restrict__ hbf, const __bf16* __restrict__ p1t,
                       const float* __restrict__ p1bias, const float* __restrict__ p1g,
                       const float* __restrict__ p1beta, const float* __restrict__ p2W,
                       const float* __restrict__ p2b, float* __restrict__ out) {
  int lane = threadIdx.x & 31, w = threadIdx.x >> 5;
  int row0 = blockIdx.x * 128 + w * 16;
  int arow = row0 + (lane & 15);
  if (arow >= N_NODES) arow = N_NODES - 1;
  const __bf16* ap = hbf + (size_t)arow * HID;
  int colb = lane & 15, half = (lane & 16) ? 8 : 0;

  v8f acc[4];
#pragma unroll
  for (int ct = 0; ct < 4; ++ct) acc[ct] = zero8();
#pragma unroll
  for (int ks = 0; ks < 4; ++ks) {
    v16bf a = frag_a_bf(ap, lane, ks * 32);
#pragma unroll
    for (int ct = 0; ct < 4; ++ct) {
      int col = 16 * ct + colb;
      acc[ct] = wmma_bf(a, frag_b_bf(p1t + (size_t)col * 128, lane, ks * 32), acc[ct]);
    }
  }
  float gv[4], btv[4], w2v[4];
#pragma unroll
  for (int ct = 0; ct < 4; ++ct) {
    int col = 16 * ct + colb;
    float bb = p1bias[col];
    gv[ct] = p1g[col]; btv[ct] = p1beta[col]; w2v[ct] = p2W[col];
#pragma unroll
    for (int r = 0; r < 8; ++r) acc[ct][r] += bb;
  }
  float ob = p2b[0];
#pragma unroll
  for (int r = 0; r < 8; ++r) {
    float s = 0.f, q = 0.f;
#pragma unroll
    for (int ct = 0; ct < 4; ++ct) { float v = acc[ct][r]; s += v; q += v * v; }
    s = half16_sum(s); q = half16_sum(q);
    float mean = s * (1.f / 64.f);
    float var  = q * (1.f / 64.f) - mean * mean;
    float rstd = rsqrtf(var + EPS_LN);
    float d = 0.f;
#pragma unroll
    for (int ct = 0; ct < 4; ++ct) {
      float v = fmaxf((acc[ct][r] - mean) * rstd * gv[ct] + btv[ct], 0.f);
      d += v * w2v[ct];
    }
    d = half16_sum(d);
    int row = row0 + r + half;
    if ((lane & 15) == 0 && row < N_NODES) out[row] = d + ob;
  }
}

// ---------------------------------------------------------------------------
// Host-side launch. Input order: top-level dict insertion order, params
// flattened as a JAX pytree (dict keys sorted, tuples in order):
//  0 x | 1 he_index | 2 he_attr | 3 he_mark | 4 he_count | 5 in_W | 6 in_b |
//  per layer l (7 + 14*l): e2n.{W1,W2,b1,b2,bt1,g1}, ln_b, ln_g,
//                          n2e.{W1,W2,b1,b2,bt1,g1}
//  35 p1.W | 36 p1.b | 37 p1b(beta) | 38 p1g(gamma) | 39 p2.W | 40 p2.b
// Workspace (~131.5 MB): h f32 | hbf | aggf | aggbf | xbf | attrbf | Wt.
// ---------------------------------------------------------------------------
extern "C" void kernel_launch(void* const* d_in, const int* in_sizes, int n_in,
                              void* d_out, int out_size, void* d_ws, size_t ws_size,
                              hipStream_t stream) {
  (void)in_sizes; (void)n_in; (void)out_size; (void)ws_size;
  const float* x        = (const float*)d_in[0];
  const int*   he_index = (const int*)d_in[1];
  const int*   he_ids   = he_index + INC_N;      // row 1 of [2, INC]
  const float* he_attr  = (const float*)d_in[2];
  const float* he_mark  = (const float*)d_in[3];
  const float* he_count = (const float*)d_in[4];
  const float* in_W     = (const float*)d_in[5];
  const float* in_b     = (const float*)d_in[6];

  struct LayerP {
    const float *e2nW1, *e2nW2, *e2nB1, *e2nB2, *e2nBt1, *e2nG1;
    const float *lnB, *lnG;
    const float *n2eW1, *n2eW2, *n2eB1, *n2eB2, *n2eBt1, *n2eG1;
  } L[2];
  for (int l = 0; l < 2; ++l) {
    int b = 7 + 14 * l;
    L[l].e2nW1  = (const float*)d_in[b + 0];
    L[l].e2nW2  = (const float*)d_in[b + 1];
    L[l].e2nB1  = (const float*)d_in[b + 2];
    L[l].e2nB2  = (const float*)d_in[b + 3];
    L[l].e2nBt1 = (const float*)d_in[b + 4];
    L[l].e2nG1  = (const float*)d_in[b + 5];
    L[l].lnB    = (const float*)d_in[b + 6];
    L[l].lnG    = (const float*)d_in[b + 7];
    L[l].n2eW1  = (const float*)d_in[b + 8];
    L[l].n2eW2  = (const float*)d_in[b + 9];
    L[l].n2eB1  = (const float*)d_in[b + 10];
    L[l].n2eB2  = (const float*)d_in[b + 11];
    L[l].n2eBt1 = (const float*)d_in[b + 12];
    L[l].n2eG1  = (const float*)d_in[b + 13];
  }
  const float* p1W    = (const float*)d_in[35];
  const float* p1bias = (const float*)d_in[36];
  const float* p1beta = (const float*)d_in[37];
  const float* p1g    = (const float*)d_in[38];
  const float* p2W    = (const float*)d_in[39];
  const float* p2b    = (const float*)d_in[40];

  char* wsb = (char*)d_ws;
  float*  h      = (float*)(wsb);                                  // 51.2 MB
  __bf16* hbf    = (__bf16*)(wsb + (size_t)51200000);              // 25.6 MB
  float*  aggf   = (float*)(wsb + (size_t)76800000);               // 25.6 MB
  __bf16* aggbf  = (__bf16*)(wsb + (size_t)102400000);             // 12.8 MB
  __bf16* xbf    = (__bf16*)(wsb + (size_t)115200000);             // 12.8 MB
  __bf16* attrbf = (__bf16*)(wsb + (size_t)128000000);             //  3.2 MB
  __bf16* wt     = (__bf16*)(wsb + (size_t)131200000);             // 0.31 MB

  __bf16* inWt = wt;              // [128][64]
  __bf16* p1t  = wt + 8192;       // [64][128]
  __bf16* n2eW1t[2], *n2eW2t[2], *e2nW1t[2], *e2nW2t[2];
  for (int l = 0; l < 2; ++l) {
    __bf16* lb = wt + 16384 + (size_t)l * 69632;
    n2eW1t[l] = lb;               // [128][128] (first 128 rows of W1)
    n2eW2t[l] = lb + 16384;       // [128][128]
    e2nW1t[l] = lb + 32768;       // [128][160]
    e2nW2t[l] = lb + 53248;       // [128][128]
  }

  const int TPB = 256;
  auto blk = [](int n) { return (n + 255) / 256; };

  // --- input / weight conversions to bf16 ---
  k_cvt_bf16<<<blk(N_NODES * NODE_DIM), TPB, 0, stream>>>(x, xbf, N_NODES * NODE_DIM);
  k_cvt_bf16<<<blk(N_EDGES * HEDGE_DIM), TPB, 0, stream>>>(he_attr, attrbf,
                                                           N_EDGES * HEDGE_DIM);
  k_transpose_bf16<<<blk(64 * 128), TPB, 0, stream>>>(in_W, inWt, 64, 128);
  k_transpose_bf16<<<blk(128 * 64), TPB, 0, stream>>>(p1W, p1t, 128, 64);
  for (int l = 0; l < 2; ++l) {
    k_transpose_bf16<<<blk(128 * 128), TPB, 0, stream>>>(L[l].n2eW1, n2eW1t[l], 128, 128);
    k_transpose_bf16<<<blk(128 * 128), TPB, 0, stream>>>(L[l].n2eW2, n2eW2t[l], 128, 128);
    k_transpose_bf16<<<blk(160 * 128), TPB, 0, stream>>>(L[l].e2nW1, e2nW1t[l], 160, 128);
    k_transpose_bf16<<<blk(128 * 128), TPB, 0, stream>>>(L[l].e2nW2, e2nW2t[l], 128, 128);
  }

  // --- input projection ---
  const int nodeBlocks = (N_NODES + 127) / 128;   // 782
  k_in_proj<<<nodeBlocks, TPB, 0, stream>>>(xbf, inWt, in_b, h, hbf);

  // --- message-passing layers ---
  const int incBlocks = INC_N / 128;              // 6250 (exact)
  for (int l = 0; l < 2; ++l) {
    hipMemsetAsync(aggf, 0, (size_t)N_EDGES * HID * sizeof(float), stream);
    k_node2edge<<<incBlocks, TPB, 0, stream>>>(
        hbf, he_ids, he_mark, n2eW1t[l], L[l].n2eW1, L[l].n2eB1, L[l].n2eG1,
        L[l].n2eBt1, n2eW2t[l], L[l].n2eB2, aggf);
    k_agg_div<<<(N_EDGES * HID) / 256, TPB, 0, stream>>>(aggf, he_count, aggbf);
    k_edge2node<<<incBlocks, TPB, 0, stream>>>(
        attrbf, aggbf, he_ids, e2nW1t[l], L[l].e2nB1, L[l].e2nG1, L[l].e2nBt1,
        e2nW2t[l], L[l].e2nB2, L[l].lnG, L[l].lnB, h, hbf);
  }

  // --- prediction head ---
  k_head<<<nodeBlocks, TPB, 0, stream>>>(hbf, p1t, p1bias, p1g, p1beta, p2W, p2b,
                                         (float*)d_out);
}
